// Layer_FISTA_55301998903876
// MI455X (gfx1250) — matrix operand order repaired and checked by reading, not compiled, not run on hardware
//
#include <hip/hip_runtime.h>

// ---------------------------------------------------------------------------
// MI455X / gfx1250 implementation.
// GEMMs in f16 inputs + f32 accumulation via v_wmma_f32_16x16x32_f16.
//   recon :  C(256x300)  @ X[b,p](300x50)      -> patch_to_image scatter
//   xpred :  A(300x300)  @ x_prev[b,p](300x50) -> X_pred
//   xu    :  gamma(B@U) * sum_p |X|            -> X_U   (VALU, tiny)
// K padded 300->320 (10 steps of 32), N padded 50->64 (4 tiles of 16).
// Per m-tile: sched_barrier(0) pins all 20 A-fragment b128 loads ahead of the
// WMMA chain -> one load clause, one wait, 10 dense accumulator-chained WMMAs.
// ---------------------------------------------------------------------------

typedef _Float16 v16h __attribute__((ext_vector_type(16)));
typedef _Float16 v8h  __attribute__((ext_vector_type(8)));
typedef float    v8f  __attribute__((ext_vector_type(8)));

#define LDS_ROW 328   // 320 K-halves + 8 pad; 328*2 = 656 B = 41*16 -> b128-aligned rows
#define KSTEPS  10    // 320 / 32

__device__ __forceinline__ v16h cat8(v8h lo, v8h hi) {
  return __builtin_shufflevector(lo, hi, 0,1,2,3,4,5,6,7,8,9,10,11,12,13,14,15);
}

// --- weight conversion / padding: C -> Cf16[256][320], A -> Af16[304][320] ---
__global__ void convert_weights(const float* __restrict__ C, const float* __restrict__ A,
                                _Float16* __restrict__ Cf, _Float16* __restrict__ Af) {
  int idx = blockIdx.x * 256 + threadIdx.x;
  if (idx < 256 * 320) {
    int r = idx / 320, c = idx - r * 320;
    Cf[idx] = (c < 300) ? (_Float16)C[r * 300 + c] : (_Float16)0.f;
  } else {
    int j = idx - 256 * 320;
    if (j < 304 * 320) {
      int r = j / 320, c = j - r * 320;
      Af[j] = (r < 300 && c < 300) ? (_Float16)A[r * 300 + c] : (_Float16)0.f;
    }
  }
}

// --- recon GEMM + patch_to_image scatter -----------------------------------
__global__ __launch_bounds__(256)
void recon_kernel(const float* __restrict__ X, const _Float16* __restrict__ Cf,
                  float* __restrict__ out) {
  __shared__ __attribute__((aligned(16))) _Float16 sX[64 * LDS_ROW];
  const int p = blockIdx.x, b = blockIdx.y;
  const int tid = threadIdx.x;
  const size_t xbase = (size_t)(b * 64 + p) * 300 * 50;

  // Stage X[b,p] (300x50 f32) into LDS as f16, transposed to [t][j], zero-padded.
  for (int idx = tid; idx < 320 * 64; idx += 256) {
    int j = idx >> 6, t = idx & 63;            // t fastest -> coalesced global reads
    float v = 0.f;
    if (j < 300 && t < 50) v = X[xbase + (size_t)j * 50 + t];
    sX[t * LDS_ROW + j] = (_Float16)v;
  }
  __syncthreads();

  const int lane   = tid & 31;
  const int w      = tid >> 5;     // wave id 0..7
  const int ntile  = w & 3;        // which 16-wide t tile
  const int mstart = w >> 2;       // 0 or 1; m += 2
  const int lrow   = lane & 15;
  const int lhi    = lane >> 4;

  // Preload all B fragments (X side) for this wave's N tile.
  // B layout: lane -> N = lane%16; halves 0-7 -> K=k0+8h+0..7, 8-15 -> K=k0+16+8h+0..7
  v16h bfr[KSTEPS];
  #pragma unroll
  for (int k = 0; k < KSTEPS; ++k) {
    const _Float16* bp = &sX[(ntile * 16 + lrow) * LDS_ROW + k * 32 + 8 * lhi];
    bfr[k] = cat8(*(const v8h*)bp, *(const v8h*)(bp + 16));
  }

  const int t = ntile * 16 + lrow;
  for (int m = mstart; m < 16; m += 2) {       // 16 M tiles (d = 0..255)
    // A layout: lane -> M = lane%16; halves 0-7 -> K=k0+8h+0..7, 8-15 -> K=k0+16+8h+0..7
    const _Float16* arow = Cf + (size_t)(m * 16 + lrow) * 320 + 8 * lhi;
    v16h afr[KSTEPS];
    #pragma unroll
    for (int k = 0; k < KSTEPS; ++k) {
      const _Float16* ap = arow + k * 32;
      afr[k] = cat8(*(const v8h*)ap, *(const v8h*)(ap + 16));
    }
    // Pin all loads before the math: one clause + one wait, fragments stay live.
    __builtin_amdgcn_sched_barrier(0);
    v8f acc = {};
    #pragma unroll
    for (int k = 0; k < KSTEPS; ++k)           // dense WMMA chain (acc RAW only)
      acc = __builtin_amdgcn_wmma_f32_16x16x32_f16(false, afr[k], false, bfr[k],
                                                   (short)0, acc, false, false);
    if (t < 50) {
      #pragma unroll
      for (int r = 0; r < 8; ++r) {
        int d = m * 16 + r + 8 * lhi;                  // pixel within patch
        int H = (p >> 3) * 16 + (d >> 4);
        int W = (p & 7) * 16 + (d & 15);
        out[((size_t)b * 16384 + H * 128 + W) * 50 + t] = acc[r];
      }
    }
  }
}

// --- X_pred GEMM -----------------------------------------------------------
__global__ __launch_bounds__(256)
void xpred_kernel(const float* __restrict__ X, const float* __restrict__ x0,
                  const _Float16* __restrict__ Af, float* __restrict__ out) {
  __shared__ __attribute__((aligned(16))) _Float16 sX[64 * LDS_ROW];
  const int p = blockIdx.x, b = blockIdx.y;
  const int tid = threadIdx.x;
  const size_t slab = (size_t)(b * 64 + p) * 300;

  // Stage x_prev[b,p] = [x0, X[...,:-1]] into LDS as f16 [t][j].
  for (int idx = tid; idx < 320 * 64; idx += 256) {
    int j = idx >> 6, t = idx & 63;
    float v = 0.f;
    if (j < 300 && t < 50)
      v = (t == 0) ? x0[slab + j] : X[(slab + j) * 50 + (t - 1)];
    sX[t * LDS_ROW + j] = (_Float16)v;
  }
  __syncthreads();

  const int lane   = tid & 31;
  const int w      = tid >> 5;
  const int ntile  = w & 3;
  const int mstart = w >> 2;
  const int lrow   = lane & 15;
  const int lhi    = lane >> 4;

  v16h bfr[KSTEPS];
  #pragma unroll
  for (int k = 0; k < KSTEPS; ++k) {
    const _Float16* bp = &sX[(ntile * 16 + lrow) * LDS_ROW + k * 32 + 8 * lhi];
    bfr[k] = cat8(*(const v8h*)bp, *(const v8h*)(bp + 16));
  }

  const int t = ntile * 16 + lrow;
  for (int m = mstart; m < 19; m += 2) {       // 19 M tiles cover i = 0..303 (pad)
    const _Float16* arow = Af + (size_t)(m * 16 + lrow) * 320 + 8 * lhi;
    v16h afr[KSTEPS];
    #pragma unroll
    for (int k = 0; k < KSTEPS; ++k) {
      const _Float16* ap = arow + k * 32;
      afr[k] = cat8(*(const v8h*)ap, *(const v8h*)(ap + 16));
    }
    __builtin_amdgcn_sched_barrier(0);
    v8f acc = {};
    #pragma unroll
    for (int k = 0; k < KSTEPS; ++k)
      acc = __builtin_amdgcn_wmma_f32_16x16x32_f16(false, afr[k], false, bfr[k],
                                                   (short)0, acc, false, false);
    if (t < 50) {
      #pragma unroll
      for (int r = 0; r < 8; ++r) {
        int i = m * 16 + r + 8 * lhi;
        if (i < 300) out[(slab + i) * 50 + t] = acc[r];
      }
    }
  }
}

// --- X_U = gamma(B@U) * sum_p |X| ------------------------------------------
__global__ __launch_bounds__(256)
void xu_kernel(const float* __restrict__ X, const float* __restrict__ U,
               const float* __restrict__ B, float* __restrict__ out) {
  int idx = blockIdx.x * 256 + threadIdx.x;
  if (idx >= 32 * 300 * 50) return;
  int t = idx % 50;
  int i = (idx / 50) % 300;
  int b = idx / 15000;
  float bu = 0.f;
  #pragma unroll 8
  for (int q = 0; q < 40; ++q) bu += B[i * 40 + q] * U[b * 2000 + q * 50 + t];
  float gamma = 0.5f * (1.0f + __expf(-bu));   // GAMMA0 = 1.0
  float s = 0.f;
  #pragma unroll 8
  for (int pp = 0; pp < 64; ++pp)
    s += fabsf(X[(size_t)b * 960000 + (size_t)pp * 15000 + (size_t)i * 50 + t]);
  out[idx] = gamma * s;
}

// ---------------------------------------------------------------------------
extern "C" void kernel_launch(void* const* d_in, const int* in_sizes, int n_in,
                              void* d_out, int out_size, void* d_ws, size_t ws_size,
                              hipStream_t stream) {
  const float* X  = (const float*)d_in[0];   // (32,64,300,1,50)
  const float* U  = (const float*)d_in[1];   // (32,1,40,1,50)
  const float* x0 = (const float*)d_in[2];   // (32,64,300,1)
  const float* A  = (const float*)d_in[3];   // (300,300)
  const float* B  = (const float*)d_in[4];   // (300,40)
  const float* C  = (const float*)d_in[5];   // (256,300)
  float* out = (float*)d_out;

  _Float16* Cf = (_Float16*)d_ws;                              // 256*320 halves
  _Float16* Af = (_Float16*)((char*)d_ws + 256 * 320 * 2);     // 304*320 halves

  convert_weights<<<700, 256, 0, stream>>>(C, A, Cf, Af);
  recon_kernel<<<dim3(64, 32), 256, 0, stream>>>(X, Cf, out);
  xpred_kernel<<<dim3(64, 32), 256, 0, stream>>>(X, x0, Af, out + 26214400);
  xu_kernel<<<1875, 256, 0, stream>>>(X, U, B, out + 56934400);
}
